// MRIC_46248207844100
// MI455X (gfx1250) — compile-verified, matrix operand order checked by reading, not run on hardware
//
#include <hip/hip_runtime.h>
#include <hip/hip_bf16.h>

// Problem constants (from reference): x1,x2 are (16384, 1024) fp32; b = 4096.
#define CDIM 1024
#define BGRP 4096

typedef __attribute__((ext_vector_type(8)))  float  v8f;
typedef __attribute__((ext_vector_type(16))) __bf16 v16bf;

union FragBF { v16bf v; uint4 q[2]; };   // 32 bytes: 16 bf16 halves
union AccF   { v8f   v; float f[8]; };   // 32 bytes: 16x16 f32 C/D slice per lane

__device__ __forceinline__ unsigned short f32_to_bf16(float f) {
    unsigned u = __float_as_uint(f);
    u += 0x7FFFu + ((u >> 16) & 1u);     // round-to-nearest-even
    return (unsigned short)(u >> 16);
}

// Block-wide sum of n per-thread partials (256 threads = 8 wave32). lds >= n*8.
__device__ __forceinline__ void block_reduce(float* v, int n, float* lds) {
    const int lane = threadIdx.x & 31;
    const int wave = threadIdx.x >> 5;
    for (int k = 0; k < n; ++k) {
        float x = v[k];
        #pragma unroll
        for (int m = 16; m >= 1; m >>= 1) x += __shfl_xor(x, m, 32);
        if (lane == 0) lds[k * 8 + wave] = x;
    }
    __syncthreads();
    for (int k = 0; k < n; ++k) {
        float s = 0.f;
        #pragma unroll
        for (int w = 0; w < 8; ++w) s += lds[k * 8 + w];
        v[k] = s;
    }
    __syncthreads();
}

__global__ void zero_kernel(float* __restrict__ p, int n) {
    int i = blockIdx.x * 256 + threadIdx.x;
    if (i < n) p[i] = 0.f;
}

// One block per group g: normalize 4 rows, attention-pool, normalize pooled,
// emit fp32 to d_out and bf16 to workspace; accumulate center-loss term.
__global__ void pool_kernel(const float* __restrict__ x1,
                            const float* __restrict__ x2,
                            float* __restrict__ out,
                            unsigned short* __restrict__ p1b,
                            unsigned short* __restrict__ p2b,
                            float* __restrict__ accum) {
    __shared__ float lds[64];
    const int g = blockIdx.x;
    const int tid = threadIdx.x;
    float4 q[2];

    for (int ii = 0; ii < 2; ++ii) {
        const float* x = (ii ? x2 : x1) + (size_t)g * (4 * CDIM);
        float4 v[4];
        float red[4];
        #pragma unroll
        for (int r = 0; r < 4; ++r) {
            v[r] = ((const float4*)(x + r * CDIM))[tid];
            red[r] = v[r].x * v[r].x + v[r].y * v[r].y + v[r].z * v[r].z + v[r].w * v[r].w;
        }
        block_reduce(red, 4, lds);
        #pragma unroll
        for (int r = 0; r < 4; ++r) {
            float s = 1.f / fmaxf(sqrtf(red[r]), 1e-12f);
            v[r].x *= s; v[r].y *= s; v[r].z *= s; v[r].w *= s;
        }
        // s_i = y_i . (sum_j y_j)
        float4 t;
        t.x = v[0].x + v[1].x + v[2].x + v[3].x;
        t.y = v[0].y + v[1].y + v[2].y + v[3].y;
        t.z = v[0].z + v[1].z + v[2].z + v[3].z;
        t.w = v[0].w + v[1].w + v[2].w + v[3].w;
        #pragma unroll
        for (int r = 0; r < 4; ++r)
            red[r] = v[r].x * t.x + v[r].y * t.y + v[r].z * t.z + v[r].w * t.w;
        block_reduce(red, 4, lds);
        float mx = fmaxf(fmaxf(red[0], red[1]), fmaxf(red[2], red[3]));
        float w[4]; float wsum = 0.f;
        #pragma unroll
        for (int r = 0; r < 4; ++r) { w[r] = __expf(red[r] - mx); wsum += w[r]; }
        const float winv = 1.f / wsum;
        float4 p = make_float4(0.f, 0.f, 0.f, 0.f);
        #pragma unroll
        for (int r = 0; r < 4; ++r) {
            float ww = w[r] * winv;
            p.x += ww * v[r].x; p.y += ww * v[r].y; p.z += ww * v[r].z; p.w += ww * v[r].w;
        }
        float pr[1] = { p.x * p.x + p.y * p.y + p.z * p.z + p.w * p.w };
        block_reduce(pr, 1, lds);
        float ps = 1.f / fmaxf(sqrtf(pr[0]), 1e-12f);
        p.x *= ps; p.y *= ps; p.z *= ps; p.w *= ps;

        // fp32 result (d_out+1 is 4B-aligned only -> scalar stores)
        float* po = out + 1 + (size_t)ii * ((size_t)BGRP * CDIM) + (size_t)g * CDIM + tid * 4;
        po[0] = p.x; po[1] = p.y; po[2] = p.z; po[3] = p.w;

        // bf16 copy for the WMMA GEMM (8B-aligned packed store)
        unsigned short* pb = (ii ? p2b : p1b) + (size_t)g * CDIM + tid * 4;
        unsigned lo = (unsigned)f32_to_bf16(p.x) | ((unsigned)f32_to_bf16(p.y) << 16);
        unsigned hi = (unsigned)f32_to_bf16(p.z) | ((unsigned)f32_to_bf16(p.w) << 16);
        *(uint2*)pb = make_uint2(lo, hi);
        q[ii] = p;
    }
    float dx = q[0].x - q[1].x, dy = q[0].y - q[1].y;
    float dz = q[0].z - q[1].z, dw = q[0].w - q[1].w;
    float cd[1] = { dx * dx + dy * dy + dz * dz + dw * dw };
    block_reduce(cd, 1, lds);
    if (tid == 0) atomicAdd(accum, sqrtf(cd[0]) * (1.0f / (float)BGRP));
}

// 128x128 sim tile per block (8 wave32, each 32x64 = 2x4 WMMA 16x16 accums),
// K=1024 in steps of 32 via v_wmma_f32_16x16x32_bf16. Fragments load directly
// from K-major bf16 memory: per-lane data = two contiguous 16B chunks.
__global__ void sim_kernel(const unsigned short* __restrict__ p1b,
                           const unsigned short* __restrict__ p2b,
                           float* __restrict__ rowsum,
                           float* __restrict__ colsum,
                           float* __restrict__ diag) {
    const int br = blockIdx.x >> 5;
    const int bc = blockIdx.x & 31;
    const int tid  = threadIdx.x;
    const int lane = tid & 31;
    const int wv   = tid >> 5;
    const int wr = wv >> 1, wc = wv & 1;
    const int rowBase = br * 128 + wr * 32;
    const int colBase = bc * 128 + wc * 64;
    const int lm   = lane & 15;
    const int hi8  = (lane >> 4) & 1;      // lanes 16..31 handle +8 rows / +8 K
    const int laneoff = hi8 * 8;

    AccF acc[2][4];
    const v8f vz = {0.f, 0.f, 0.f, 0.f, 0.f, 0.f, 0.f, 0.f};
    #pragma unroll
    for (int mi = 0; mi < 2; ++mi)
        #pragma unroll
        for (int ni = 0; ni < 4; ++ni) acc[mi][ni].v = vz;

    for (int kt = 0; kt < CDIM / 32; ++kt) {
        const int kb = kt * 32 + laneoff;
        FragBF A[2], Bf[4];
        #pragma unroll
        for (int mi = 0; mi < 2; ++mi) {
            const uint4* ap = (const uint4*)(p1b + (size_t)(rowBase + mi * 16 + lm) * CDIM + kb);
            A[mi].q[0] = ap[0];            // K = kb .. kb+7
            A[mi].q[1] = ap[2];            // K = kb+16 .. kb+23
        }
        #pragma unroll
        for (int ni = 0; ni < 4; ++ni) {
            const uint4* bp = (const uint4*)(p2b + (size_t)(colBase + ni * 16 + lm) * CDIM + kb);
            Bf[ni].q[0] = bp[0];
            Bf[ni].q[1] = bp[2];
        }
        // WGP-scope prefetch of next K slab (locality 3 -> near caches). A
        // one-slab overrun past the bf16 arrays stays inside the ws allocation.
        __builtin_prefetch(p1b + (size_t)(rowBase + lm) * CDIM + kb + 32, 0, 3);
        __builtin_prefetch(p2b + (size_t)(colBase + lm) * CDIM + kb + 32, 0, 3);
        #pragma unroll
        for (int mi = 0; mi < 2; ++mi)
            #pragma unroll
            for (int ni = 0; ni < 4; ++ni)
                acc[mi][ni].v = __builtin_amdgcn_wmma_f32_16x16x32_bf16(
                    false, A[mi].v, false, Bf[ni].v,
                    (short)0, acc[mi][ni].v, false, false);
    }

    // Epilogue: sim entries are unit-vector dots (<= 1), so use exp(s - 1).
    // Row partial sums: reduce over 16 lanes of each half-wave, atomic into rowsum.
    #pragma unroll
    for (int mi = 0; mi < 2; ++mi) {
        float es[8];
        #pragma unroll
        for (int r = 0; r < 8; ++r) es[r] = 0.f;
        #pragma unroll
        for (int ni = 0; ni < 4; ++ni)
            #pragma unroll
            for (int r = 0; r < 8; ++r)
                es[r] += __expf(acc[mi][ni].f[r] - 1.0f);
        #pragma unroll
        for (int r = 0; r < 8; ++r) {
            float s = es[r];
            s += __shfl_xor(s, 1, 32);
            s += __shfl_xor(s, 2, 32);
            s += __shfl_xor(s, 4, 32);
            s += __shfl_xor(s, 8, 32);
            if (lm == 0) atomicAdd(&rowsum[rowBase + mi * 16 + hi8 * 8 + r], s);
        }
    }
    // Column partial sums: per lane column n = colBase + ni*16 + lm; fold halves.
    #pragma unroll
    for (int ni = 0; ni < 4; ++ni) {
        float cs = 0.f;
        #pragma unroll
        for (int mi = 0; mi < 2; ++mi)
            #pragma unroll
            for (int r = 0; r < 8; ++r)
                cs += __expf(acc[mi][ni].f[r] - 1.0f);
        cs += __shfl_xor(cs, 16, 32);
        if (hi8 == 0) atomicAdd(&colsum[colBase + ni * 16 + lm], cs);
    }
    // Diagonal: each (i,i) hit by exactly one lane/vgpr across the grid.
    #pragma unroll
    for (int mi = 0; mi < 2; ++mi)
        #pragma unroll
        for (int ni = 0; ni < 4; ++ni)
            #pragma unroll
            for (int r = 0; r < 8; ++r) {
                int R  = rowBase + mi * 16 + hi8 * 8 + r;
                int Cc = colBase + ni * 16 + lm;
                if (R == Cc) diag[R] = acc[mi][ni].f[r];
            }
}

__global__ void finalize_kernel(const float* __restrict__ rowsum,
                                const float* __restrict__ colsum,
                                const float* __restrict__ diag,
                                const float* __restrict__ accum,
                                float* __restrict__ out) {
    __shared__ float lds[16];
    const int tid = threadIdx.x;
    float lt = 0.f, li = 0.f;
    for (int i = tid; i < BGRP; i += 256) {
        float d = diag[i];
        lt += 1.0f + __logf(rowsum[i]) - d;   // lse_row - sim[i,i]
        li += 1.0f + __logf(colsum[i]) - d;   // lse_col - sim[i,i]
    }
    float v[2] = { lt, li };
    block_reduce(v, 2, lds);
    if (tid == 0) out[0] = (v[0] + v[1]) * (1.0f / (float)BGRP) + accum[0];
}

extern "C" void kernel_launch(void* const* d_in, const int* in_sizes, int n_in,
                              void* d_out, int out_size, void* d_ws, size_t ws_size,
                              hipStream_t stream) {
    (void)in_sizes; (void)n_in; (void)out_size; (void)ws_size;
    const float* x1 = (const float*)d_in[0];
    const float* x2 = (const float*)d_in[1];
    float* out = (float*)d_out;   // [loss, p1(4096x1024), p2(4096x1024)]

    // Workspace layout (needs ~16.1 MB):
    //   [0, 8MB):       p1 bf16   (4096x1024 u16)
    //   [8MB, 16MB):    p2 bf16
    //   [16MB, ...):    rowsum[4096], colsum[4096], diag[4096], accum[4]  (f32)
    char* ws = (char*)d_ws;
    unsigned short* p1b = (unsigned short*)ws;
    unsigned short* p2b = (unsigned short*)(ws + (size_t)BGRP * CDIM * 2);
    float* rowsum = (float*)(ws + (size_t)2 * BGRP * CDIM * 2);
    float* colsum = rowsum + BGRP;
    float* diag   = colsum + BGRP;
    float* accum  = diag + BGRP;

    const int nzero = 3 * BGRP + 4;  // rowsum, colsum, diag, accum
    zero_kernel<<<(nzero + 255) / 256, 256, 0, stream>>>(rowsum, nzero);
    pool_kernel<<<BGRP, 256, 0, stream>>>(x1, x2, out, p1b, p2b, accum);
    sim_kernel<<<(BGRP / 128) * (BGRP / 128), 256, 0, stream>>>(p1b, p2b, rowsum, colsum, diag);
    finalize_kernel<<<1, 256, 0, stream>>>(rowsum, colsum, diag, accum, out);
}
